// TsallisRouter_73478300500466
// MI455X (gfx1250) — matrix operand order repaired and checked by reading, not compiled, not run on hardware
//
#include <hip/hip_runtime.h>

// ---------------------------------------------------------------------------
// TsallisRouter (q=2 -> sparsemax) for gfx1250 (MI455X, wave32).
//
//   h         = relu(x @ w1 + b1)        [8192 x 256]
//   utilities = h @ w2 + b2              [8192 x 64]
//   tau via 50-iter bisection per row; p = relu(u - tau) / (sum + 1e-8)
//
// Precision: split-bf16 (hi+lo) emulated fp32 on v_wmma_f32_16x16x32_bf16.
// Weights are pre-converted & pre-swizzled into WMMA B-fragment order once;
// the hot loop is b128 loads + pass-major WMMA bursts (no hazard NOPs, no
// per-iteration cvt VALU).
// ---------------------------------------------------------------------------

#define BDIM 128

constexpr int Bm = 16;      // M rows per block
constexpr int Dk = 4096;    // GEMM1 inner dim
constexpr int Hn = 256;     // hidden width
constexpr int En = 64;      // experts
constexpr int KC = 32;      // WMMA bf16 K

typedef __attribute__((ext_vector_type(16))) __bf16 v16bf;
typedef __attribute__((ext_vector_type(8)))  __bf16 v8bf;
typedef __attribute__((ext_vector_type(4)))  __bf16 v4bf;
typedef __attribute__((ext_vector_type(8)))  float  v8f;

// ---------------------------------------------------------------------------
// Prep: swizzle fp32 weight [K x N] (row-major) into WMMA B-fragment order,
// split into bf16 hi/lo. Fragment order: o = ((kc*nTiles + nt)*32 + lane)*16+e
// with B-layout (32x16, 16-bit): elem e -> K = kc*32 + (e%8) + 16*(e/8) + 8*(lane/16),
//                                         N = nt*16 + lane%16.
// Each lane's 16 bf16 land contiguous (32B) -> main loop loads them as b128s.
// ---------------------------------------------------------------------------
__global__ void swizzle_b_kernel(const float* __restrict__ w,
                                 __bf16* __restrict__ dhi,
                                 __bf16* __restrict__ dlo,
                                 int N, int nTiles, int total)
{
    const int o = blockIdx.x * blockDim.x + threadIdx.x;
    if (o >= total) return;
    const int e    = o & 15;
    const int lane = (o >> 4) & 31;
    const int nt   = (o >> 9) % nTiles;
    const int kc   = (o >> 9) / nTiles;
    const int k = kc * KC + (e & 7) + ((e >> 3) << 4) + ((lane >> 4) << 3);
    const int n = nt * 16 + (lane & 15);
    const float  f = w[(size_t)k * N + n];
    const __bf16 h = (__bf16)f;
    dhi[o] = h;
    dlo[o] = (__bf16)(f - (float)h);
}

// ---------------------------------------------------------------------------
// Fused main kernel: 512 blocks x 128 threads (4 waves); block owns 16 rows.
// launch_bounds(...,1) relaxes VGPR pressure so all 8 B-fragments stay live.
// ---------------------------------------------------------------------------
__global__ __launch_bounds__(BDIM, 1)
void tsallis_router_kernel(const float*  __restrict__ x,
                           const __bf16* __restrict__ w1h,
                           const __bf16* __restrict__ w1l,
                           const float*  __restrict__ b1,
                           const __bf16* __restrict__ w2h,
                           const __bf16* __restrict__ w2l,
                           const float*  __restrict__ b2,
                           float* __restrict__ out)
{
    // Split-bf16 staging buffers. Row strides 80B / 528B are 16B-aligned and
    // give conflict-free half-wave row access.
    __shared__ __bf16 lds_xh[Bm][KC + 8];
    __shared__ __bf16 lds_xl[Bm][KC + 8];
    __shared__ __bf16 lds_hh[Bm][Hn + 8];
    __shared__ __bf16 lds_hl[Bm][Hn + 8];
    __shared__ float  lds_u [Bm][En + 2];

    const int tid  = threadIdx.x;
    const int wave = tid >> 5;             // 0..3
    const int lane = tid & 31;
    const int m    = lane & 15;
    const int half = lane >> 4;
    const int row0 = blockIdx.x * Bm;

    // ------------------- GEMM1: 16 x 256 tile of h -------------------------
    v8f acc[4] = {};                       // wave owns N-tiles 4*wave .. 4*wave+3

    #pragma unroll 1
    for (int k0 = 0; k0 < Dk; k0 += KC) {
        const int kc = k0 >> 5;

        // Stage + convert x[row0..+15][k0..+31]: one float4/thread -> hi/lo bf16
        {
            const int r  = tid >> 3;               // 0..15
            const int c4 = (tid & 7) << 2;         // 0,4,...,28
            const float4 v = *(const float4*)(x + (size_t)(row0 + r) * Dk + k0 + c4);
            const __bf16 h0 = (__bf16)v.x, h1 = (__bf16)v.y,
                         h2 = (__bf16)v.z, h3 = (__bf16)v.w;
            v4bf vh = {h0, h1, h2, h3};
            v4bf vl = {(__bf16)(v.x - (float)h0), (__bf16)(v.y - (float)h1),
                       (__bf16)(v.z - (float)h2), (__bf16)(v.w - (float)h3)};
            *(v4bf*)&lds_xh[r][c4] = vh;           // 8B stores, aligned
            *(v4bf*)&lds_xl[r][c4] = vl;
        }
        __syncthreads();

        // A fragments (16x32 bf16): elem e -> K = (e%8) + 16*(e/8) + 8*half
        // => two contiguous 8-elem runs per lane: b128 LDS loads.
        const v8bf ah0 = *(const v8bf*)&lds_xh[m][(half << 3)];
        const v8bf ah1 = *(const v8bf*)&lds_xh[m][(half << 3) + 16];
        const v8bf al0 = *(const v8bf*)&lds_xl[m][(half << 3)];
        const v8bf al1 = *(const v8bf*)&lds_xl[m][(half << 3) + 16];
        const v16bf a_hi = __builtin_shufflevector(ah0, ah1, 0,1,2,3,4,5,6,7,8,9,10,11,12,13,14,15);
        const v16bf a_lo = __builtin_shufflevector(al0, al1, 0,1,2,3,4,5,6,7,8,9,10,11,12,13,14,15);

        // Load ALL 8 B fragments up front (pre-swizzled 32B vector loads).
        v16bf bh[4], bl[4];
        #pragma unroll
        for (int t = 0; t < 4; ++t) {
            const size_t base = (((size_t)kc * 16 + wave * 4 + t) * 32 + lane) * 16;
            bh[t] = *(const v16bf*)(w1h + base);
            bl[t] = *(const v16bf*)(w1l + base);
        }

        // Pass order hh -> lh -> hl: bh[] used in the first two passes so the
        // bl[] loads get 8 WMMAs of slack to land; consecutive WMMAs hit
        // different accumulators -> no RAW chains, no hazard NOPs.
        #pragma unroll
        for (int t = 0; t < 4; ++t)
            acc[t] = __builtin_amdgcn_wmma_f32_16x16x32_bf16(false, a_hi, false, bh[t], (short)0, acc[t], false, false);
        #pragma unroll
        for (int t = 0; t < 4; ++t)
            acc[t] = __builtin_amdgcn_wmma_f32_16x16x32_bf16(false, a_lo, false, bh[t], (short)0, acc[t], false, false);
        #pragma unroll
        for (int t = 0; t < 4; ++t)
            acc[t] = __builtin_amdgcn_wmma_f32_16x16x32_bf16(false, a_hi, false, bl[t], (short)0, acc[t], false, false);

        __syncthreads();

        // Prefetch next x chunk while WMMAs drain.
        if (k0 + KC < Dk) {
            __builtin_prefetch(x + (size_t)(row0 + (tid >> 3)) * Dk + k0 + KC + ((tid & 7) << 2), 0, 0);
        }
    }

    // bias + relu; C layout: vgpr v -> row = v + 8*half, col = n. Park h as
    // split-bf16 in LDS so GEMM2 A-fragments are pure b128 loads.
    #pragma unroll
    for (int t = 0; t < 4; ++t) {
        const int n = wave * 64 + t * 16 + m;
        const float bias = b1[n];
        #pragma unroll
        for (int v = 0; v < 8; ++v) {
            const int r = v + (half << 3);
            const float  hv = fmaxf(acc[t][v] + bias, 0.0f);
            const __bf16 hb = (__bf16)hv;
            lds_hh[r][n] = hb;
            lds_hl[r][n] = (__bf16)(hv - (float)hb);
        }
    }
    __syncthreads();

    // ------------------- GEMM2: utilities = h @ w2 + b2 --------------------
    // Two accumulators (even/odd K chunks) break the serial WMMA chain.
    v8f acc2a = {}, acc2b = {};
    #pragma unroll 1
    for (int kc = 0; kc < Hn / KC; kc += 2) {
        v16bf a_hi[2], a_lo[2], b_hi[2], b_lo[2];
        #pragma unroll
        for (int u = 0; u < 2; ++u) {
            const int kb = (kc + u) * KC;
            const v8bf ah0 = *(const v8bf*)&lds_hh[m][kb + (half << 3)];
            const v8bf ah1 = *(const v8bf*)&lds_hh[m][kb + (half << 3) + 16];
            const v8bf al0 = *(const v8bf*)&lds_hl[m][kb + (half << 3)];
            const v8bf al1 = *(const v8bf*)&lds_hl[m][kb + (half << 3) + 16];
            a_hi[u] = __builtin_shufflevector(ah0, ah1, 0,1,2,3,4,5,6,7,8,9,10,11,12,13,14,15);
            a_lo[u] = __builtin_shufflevector(al0, al1, 0,1,2,3,4,5,6,7,8,9,10,11,12,13,14,15);
            const size_t base = (((size_t)(kc + u) * 4 + wave) * 32 + lane) * 16;
            b_hi[u] = *(const v16bf*)(w2h + base);
            b_lo[u] = *(const v16bf*)(w2l + base);
        }
        acc2a = __builtin_amdgcn_wmma_f32_16x16x32_bf16(false, a_hi[0], false, b_hi[0], (short)0, acc2a, false, false);
        acc2b = __builtin_amdgcn_wmma_f32_16x16x32_bf16(false, a_hi[1], false, b_hi[1], (short)0, acc2b, false, false);
        acc2a = __builtin_amdgcn_wmma_f32_16x16x32_bf16(false, a_lo[0], false, b_hi[0], (short)0, acc2a, false, false);
        acc2b = __builtin_amdgcn_wmma_f32_16x16x32_bf16(false, a_lo[1], false, b_hi[1], (short)0, acc2b, false, false);
        acc2a = __builtin_amdgcn_wmma_f32_16x16x32_bf16(false, a_hi[0], false, b_lo[0], (short)0, acc2a, false, false);
        acc2b = __builtin_amdgcn_wmma_f32_16x16x32_bf16(false, a_hi[1], false, b_lo[1], (short)0, acc2b, false, false);
    }
    {
        const float bias = b2[wave * 16 + m];
        #pragma unroll
        for (int v = 0; v < 8; ++v)
            lds_u[v + (half << 3)][wave * 16 + m] = (acc2a[v] + acc2b[v]) + bias;
    }
    __syncthreads();

    // ------------- sparsemax (q=2 -> EXP=1, no powf) per row ---------------
    // All 128 threads: 8 threads per row, 8 utilities each in registers;
    // width-8 xor-butterfly reductions (intra-wave on wave32) keep all lanes
    // of a group bit-identical, so the bisection branch is group-uniform.
    {
        const int r   = tid >> 3;              // row 0..15
        const int sub = tid & 7;               // lane-in-row
        float u[8];
        float lmin = 1e30f, lmax = -1e30f;
        #pragma unroll
        for (int e = 0; e < 8; ++e) {
            u[e] = lds_u[r][sub * 8 + e];
            lmin = fminf(lmin, u[e]);
            lmax = fmaxf(lmax, u[e]);
        }
        #pragma unroll
        for (int off = 1; off < 8; off <<= 1) {
            lmin = fminf(lmin, __shfl_xor(lmin, off, 8));
            lmax = fmaxf(lmax, __shfl_xor(lmax, off, 8));
        }
        float lo = lmin - 10.0f;
        float hi = lmax;
        #pragma unroll 1
        for (int it = 0; it < 50; ++it) {
            const float mid = 0.5f * (lo + hi);
            float f = 0.0f;
            #pragma unroll
            for (int e = 0; e < 8; ++e)
                f += fmaxf(u[e] - mid, 0.0f);
            #pragma unroll
            for (int off = 1; off < 8; off <<= 1)
                f += __shfl_xor(f, off, 8);
            if (f - 1.0f > 0.0f) lo = mid; else hi = mid;
        }
        const float tau = 0.5f * (lo + hi);

        float s = 0.0f;
        #pragma unroll
        for (int e = 0; e < 8; ++e)
            s += fmaxf(u[e] - tau, 0.0f);
        #pragma unroll
        for (int off = 1; off < 8; off <<= 1)
            s += __shfl_xor(s, off, 8);
        const float inv = 1.0f / (s + 1e-8f);

        float* o = out + (size_t)(row0 + r) * En + sub * 8;
        #pragma unroll
        for (int e = 0; e < 8; e += 4) {
            float4 v4;
            v4.x = fmaxf(u[e + 0] - tau, 0.0f) * inv;
            v4.y = fmaxf(u[e + 1] - tau, 0.0f) * inv;
            v4.z = fmaxf(u[e + 2] - tau, 0.0f) * inv;
            v4.w = fmaxf(u[e + 3] - tau, 0.0f) * inv;
            *(float4*)(o + e) = v4;
        }
    }
}

extern "C" void kernel_launch(void* const* d_in, const int* in_sizes, int n_in,
                              void* d_out, int out_size, void* d_ws, size_t ws_size,
                              hipStream_t stream) {
    const float* x  = (const float*)d_in[0];   // [8192, 4096]
    const float* w1 = (const float*)d_in[1];   // [4096, 256]
    const float* b1 = (const float*)d_in[2];   // [256]
    const float* w2 = (const float*)d_in[3];   // [256, 64]
    const float* b2 = (const float*)d_in[4];   // [64]
    float* out = (float*)d_out;                // [8192, 64]

    (void)in_sizes; (void)n_in; (void)out_size;

    // Workspace layout (split-bf16 swizzled weights): ~4.1 MB total.
    const int total1 = Dk * Hn;                // 1,048,576 elems
    const int total2 = Hn * En;                // 16,384 elems
    __bf16* w1h = (__bf16*)d_ws;
    __bf16* w1l = w1h + total1;
    __bf16* w2h = w1l + total1;
    __bf16* w2l = w2h + total2;
    (void)ws_size;                             // requires >= (2*total1+2*total2)*2 bytes

    // One-time (per launch) weight conversion + swizzle; stream-ordered.
    swizzle_b_kernel<<<(total1 + 255) / 256, 256, 0, stream>>>(w1, w1h, w1l, Hn, Hn / 16, total1);
    swizzle_b_kernel<<<(total2 + 255) / 256, 256, 0, stream>>>(w2, w2h, w2l, En, En / 16, total2);

    const int nblocks = 8192 / Bm;             // 512 blocks, 16 rows each
    tsallis_router_kernel<<<nblocks, BDIM, 0, stream>>>(x, w1h, w1l, b1, w2h, w2l, b2, out);
}
